// mduvsr_3478923510428
// MI455X (gfx1250) — compile-verified, matrix operand order checked by reading, not run on hardware
//
#include <hip/hip_runtime.h>
#include <hip/hip_bf16.h>

// ---------------------------------------------------------------------------
// MDUVSR forward for MI455X (gfx1250).  All dense convs -> implicit GEMM on
// v_wmma_f32_16x16x32_bf16 (wave32).  Conv weights staged block-wide in LDS
// via async global->LDS DMA (ASYNCcnt); deformable gather feeds B-fragments
// straight from VALU bilinear sampling staged via LDS.
// Workspace requirement: ~81.2 MB (see layout in kernel_launch).
// ---------------------------------------------------------------------------

#define NCc   3
#define NKc   64
#define K2c   9
#define Hc    128
#define Wc    128
#define HWc   (Hc*Wc)          // 16384
#define Bc    4
#define NPIX  (Bc*HWc)         // 65536
#define NTIL  (NPIX/16)        // 4096

typedef __attribute__((ext_vector_type(16))) __bf16 v16bf;
typedef __attribute__((ext_vector_type(8)))  float  v8f;
typedef __attribute__((ext_vector_type(4)))  int    v4i;

union BF16x16 { v16bf v; unsigned short u[16]; };

#if defined(__has_builtin)
#if __has_builtin(__builtin_amdgcn_global_load_async_to_lds_b128)
#define HAVE_ASYNC_LDS 1
#endif
#endif

static __device__ __forceinline__ unsigned short f2bf(float f) {
    unsigned int u = __float_as_uint(f);
    unsigned int r = u + 0x7FFFu + ((u >> 16) & 1u);   // round-to-nearest-even
    return (unsigned short)(r >> 16);
}
static __device__ __forceinline__ float sigm(float v) {
    return 1.f / (1.f + __expf(-v));
}
// 16-bit WMMA fragment K index for (lane, element) per CDNA5 ISA 7.12.2.
static __device__ __forceinline__ int kmap(int e, int hl) {
    return e + ((e >= 8) ? 8 : 0) + hl * 8;
}

// Cooperative block-wide copy of packed weights (global -> LDS).
static __device__ __forceinline__ void stage_weights(
        const unsigned short* __restrict__ wA, unsigned short* swA, int totalUShort) {
#if defined(HAVE_ASYNC_LDS)
    // gfx1250 async DMA into LDS, 16B per lane per op, tracked by ASYNCcnt.
    // Builtin takes int4 pointers: AS1 source, AS3 destination.
    typedef __attribute__((address_space(1))) v4i gv4i_t;
    typedef __attribute__((address_space(3))) v4i lv4i_t;
    const int n128 = totalUShort >> 3;
    for (int i = threadIdx.x; i < n128; i += 256) {
        gv4i_t* g = (gv4i_t*)(unsigned long long)(const void*)(wA + i * 8);
        lv4i_t* l = (lv4i_t*)(unsigned int)(unsigned long long)(void*)(swA + i * 8);
        __builtin_amdgcn_global_load_async_to_lds_b128(g, l, 0, 0);
    }
    asm volatile("s_wait_asynccnt 0x0" ::: "memory");
#else
    const uint4* g4 = (const uint4*)wA;
    uint4* s4 = (uint4*)swA;
    const int n128 = totalUShort >> 3;
    for (int i = threadIdx.x; i < n128; i += 256) s4[i] = g4[i];
#endif
    __syncthreads();
}

// ---------------------------------------------------------------------------
// Pack conv weights [Cout, CinW, 3, 3] (optionally two stacked tensors) into
// A-fragment order: dst[((mt*ksteps + ks)*32 + lane)*16 + e], bf16.
// ---------------------------------------------------------------------------
__global__ void pack_a(const float* __restrict__ w1, int c1,
                       const float* __restrict__ w2, int c2,
                       int cinw, int cinuse, int mtiles, int ksteps,
                       unsigned short* __restrict__ dst) {
    int t = blockIdx.x * 256 + threadIdx.x;
    int total = mtiles * ksteps * 512;
    if (t >= total) return;
    int mt = t / (ksteps * 512);
    int r  = t - mt * (ksteps * 512);
    int ks = r / 512;
    int le = r - ks * 512;
    int lane = le >> 4, e = le & 15;
    int M  = mt * 16 + (lane & 15);
    int k  = ks * 32 + kmap(e, lane >> 4);
    float v = 0.f;
    if (k < cinuse * 9) {
        int c = k / 9, tap = k - c * 9;
        if (M < c1)                      v = w1[(M * cinw + c) * 9 + tap];
        else if (w2 && M < c1 + c2)      v = w2[((M - c1) * cinw + c) * 9 + tap];
    }
    dst[t] = f2bf(v);
}

// Fill channels 64..66 (LR image) of both 67-channel activation buffers.
__global__ void set_lr(const float* __restrict__ X,
                       float* __restrict__ bufA, float* __restrict__ bufB) {
    int t = blockIdx.x * 256 + threadIdx.x;          // < 4*3*16384
    int b = t / (3 * HWc);
    int r = t - b * 3 * HWc;
    int c = r / HWc;
    int o = r - c * HWc;
    float v = X[t];
    bufA[((b * 67 + 64 + c) << 14) + o] = v;
    bufB[((b * 67 + 64 + c) << 14) + o] = v;
}

// ---------------------------------------------------------------------------
// Implicit-GEMM 3x3 conv via WMMA.  One wave per 16-pixel tile, MT M-tiles,
// KS k-steps (K = KS*32).  Weights staged in LDS block-wide.
// MODE 0: plain (+bias)         MODE 1: fused ConvLSTM (MT==16)
// MODE 2: offsets + 2*sigmoid(mask) (MT==2, Cout 18+9)
// ---------------------------------------------------------------------------
template <int MODE, int MT, int KS>
__global__ __launch_bounds__(256)
void conv_wmma(const float* __restrict__ xin, int cin,
               const unsigned short* __restrict__ wA,
               const float* __restrict__ bias0, const float* __restrict__ bias1,
               float* __restrict__ out0, float* __restrict__ out1,
               float* __restrict__ out2, int cstride, int coutvalid) {
    __shared__ __attribute__((aligned(16))) unsigned short swA[MT * KS * 512];
    stage_weights(wA, swA, MT * KS * 512);

    const int lane = threadIdx.x & 31;
    const int wave = threadIdx.x >> 5;
    const int tile = blockIdx.x * 8 + wave;
    const int p  = lane & 15, hl = lane >> 4;
    const int pix = tile * 16 + p;
    const int b  = pix >> 14;
    const int rr = pix & 16383;
    const int y  = rr >> 7;
    const int x  = rr & 127;
    const int kcnt = cin * 9;

    v8f acc[MT];
    const v8f vz = {0.f, 0.f, 0.f, 0.f, 0.f, 0.f, 0.f, 0.f};
#pragma unroll
    for (int m = 0; m < MT; ++m) acc[m] = vz;

    for (int ks = 0; ks < KS; ++ks) {
        // Hoisted LDS A-fragment loads (no dependence on the B build below).
        BF16x16 af[MT];
        const unsigned short* wp = swA + ks * 512 + lane * 16;
#pragma unroll
        for (int m = 0; m < MT; ++m)
            af[m].v = *(const v16bf*)(wp + m * KS * 512);

        BF16x16 bf;
#pragma unroll
        for (int e = 0; e < 16; ++e) {
            int k = ks * 32 + kmap(e, hl);
            float v = 0.f;
            if (k < kcnt) {
                int c = k / 9, tap = k - c * 9;
                int yy = y + tap / 3 - 1, xx = x + tap % 3 - 1;
                if ((unsigned)yy < (unsigned)Hc && (unsigned)xx < (unsigned)Wc)
                    v = xin[((b * cin + c) << 14) + (yy << 7) + xx];
            }
            bf.u[e] = f2bf(v);
        }
#pragma unroll
        for (int m = 0; m < MT; ++m)
            acc[m] = __builtin_amdgcn_wmma_f32_16x16x32_bf16(
                false, af[m].v, false, bf.v, (short)0, acc[m], false, false);
    }

    const int po = (y << 7) + x;
    if constexpr (MODE == 0) {
#pragma unroll
        for (int m = 0; m < MT; ++m)
#pragma unroll
            for (int r2 = 0; r2 < 8; ++r2) {
                int ch = m * 16 + r2 + 8 * hl;
                if (ch < coutvalid)
                    out0[((b * cstride + ch) << 14) + po] = acc[m][r2] + bias0[ch];
            }
    } else if constexpr (MODE == 1) {
        // ConvLSTM with h=c=0: cell = sig(i)*tanh(g); hid = sig(o)*tanh(cell).
#pragma unroll
        for (int q = 0; q < 4; ++q)
#pragma unroll
            for (int r2 = 0; r2 < 8; ++r2) {
                int nk = q * 16 + r2 + 8 * hl;
                float iv = acc[q][r2]      + bias0[nk];
                float ov = acc[q + 8][r2]  + bias0[nk + 128];
                float gv = acc[q + 12][r2] + bias0[nk + 192];
                float cv = sigm(iv) * tanhf(gv);
                float hv = sigm(ov) * tanhf(cv);
                out0[((b * 67 + nk) << 14) + po] = hv;   // stage buffer
                out1[((b * 64 + nk) << 14) + po] = hv;   // hid output
                out2[((b * 64 + nk) << 14) + po] = cv;   // cell output
            }
    } else {  // MODE 2: offsets (18) + modulation mask (9)
#pragma unroll
        for (int m = 0; m < MT; ++m)
#pragma unroll
            for (int r2 = 0; r2 < 8; ++r2) {
                int ch = m * 16 + r2 + 8 * hl;
                if (ch < 18)
                    out0[((b * 18 + ch) << 14) + po] = acc[m][r2] + bias0[ch];
                else if (ch < 27)
                    out1[((b * 9 + (ch - 18)) << 14) + po] =
                        2.f * sigm(acc[m][r2] + bias1[ch - 18]);
            }
    }
}

// ---------------------------------------------------------------------------
// DCNv2 core: bilinear-gathered, mask-modulated patches (K = 67*9 = 603)
// GEMM'd against packed 64x608 weights (76 KB -> stays in global; loads are
// hoisted ahead of the long bilinear B-build so they overlap it).
// ---------------------------------------------------------------------------
__global__ __launch_bounds__(256)
void deform_wmma(const float* __restrict__ xin, const float* __restrict__ off,
                 const float* __restrict__ mask,
                 const unsigned short* __restrict__ wA,
                 const float* __restrict__ bias, float* __restrict__ xout) {
    __shared__ int   sY0[8 * 144], sY1[8 * 144], sX0[8 * 144], sX1[8 * 144];
    __shared__ float sW00[8 * 144], sW01[8 * 144], sW10[8 * 144], sW11[8 * 144];

    const int lane = threadIdx.x & 31;
    const int wave = threadIdx.x >> 5;
    const int tile = blockIdx.x * 8 + wave;
    const int base = wave * 144;

    const int tpix = tile * 16;
    const int tb  = tpix >> 14;
    const int trr = tpix & 16383;
    const int ty  = trr >> 7;
    const int tx0 = trr & 127;

    for (int it = lane; it < 144; it += 32) {
        int pp = it / 9, t = it - pp * 9;
        int xx = tx0 + pp;
        int po = (ty << 7) + xx;
        float oy = off[((tb * 18 + 2 * t) << 14) + po];
        float ox = off[((tb * 18 + 2 * t + 1) << 14) + po];
        float m  = mask[((tb * 9 + t) << 14) + po];
        float py = (float)ty + (float)(t / 3 - 1) + oy;
        float px = (float)xx + (float)(t % 3 - 1) + ox;
        float fy = floorf(py), fx = floorf(px);
        float wy = py - fy, wx = px - fx;
        int iy0 = (int)fy, ix0 = (int)fx;
        int iy1 = iy0 + 1, ix1 = ix0 + 1;
        float vy0 = (iy0 >= 0 && iy0 < Hc) ? 1.f : 0.f;
        float vy1 = (iy1 >= 0 && iy1 < Hc) ? 1.f : 0.f;
        float vx0 = (ix0 >= 0 && ix0 < Wc) ? 1.f : 0.f;
        float vx1 = (ix1 >= 0 && ix1 < Wc) ? 1.f : 0.f;
        sW00[base + it] = (1.f - wy) * (1.f - wx) * m * vy0 * vx0;
        sW01[base + it] = (1.f - wy) * wx * m * vy0 * vx1;
        sW10[base + it] = wy * (1.f - wx) * m * vy1 * vx0;
        sW11[base + it] = wy * wx * m * vy1 * vx1;
        sY0[base + it] = min(max(iy0, 0), Hc - 1);
        sY1[base + it] = min(max(iy1, 0), Hc - 1);
        sX0[base + it] = min(max(ix0, 0), Wc - 1);
        sX1[base + it] = min(max(ix1, 0), Wc - 1);
    }
    __syncthreads();

    const int p  = lane & 15, hl = lane >> 4;
    const int pix = tile * 16 + p;
    const int b  = pix >> 14;
    const int rr = pix & 16383;
    const int y  = rr >> 7;
    const int x  = rr & 127;
    const float* xb = xin + ((size_t)(b * 67) << 14);

    v8f acc[4];
    const v8f vz = {0.f, 0.f, 0.f, 0.f, 0.f, 0.f, 0.f, 0.f};
#pragma unroll
    for (int m = 0; m < 4; ++m) acc[m] = vz;

    for (int ks = 0; ks < 19; ++ks) {
        // Issue all A-fragment loads first; they overlap the bilinear B build.
        BF16x16 af[4];
        const unsigned short* wp = wA + (size_t)ks * 512 + lane * 16;
#pragma unroll
        for (int m = 0; m < 4; ++m)
            af[m].v = *(const v16bf*)(wp + (size_t)m * 19 * 512);

        BF16x16 bf;
#pragma unroll
        for (int e = 0; e < 16; ++e) {
            int k = ks * 32 + kmap(e, hl);
            float v = 0.f;
            if (k < 603) {
                int c = k / 9, t = k - c * 9;
                int id = base + p * 9 + t;
                const float* xc = xb + ((size_t)c << 14);
                int y0 = sY0[id], y1 = sY1[id], x0 = sX0[id], x1 = sX1[id];
                v = sW00[id] * xc[(y0 << 7) + x0] + sW01[id] * xc[(y0 << 7) + x1]
                  + sW10[id] * xc[(y1 << 7) + x0] + sW11[id] * xc[(y1 << 7) + x1];
            }
            bf.u[e] = f2bf(v);
        }
#pragma unroll
        for (int m = 0; m < 4; ++m)
            acc[m] = __builtin_amdgcn_wmma_f32_16x16x32_bf16(
                false, af[m].v, false, bf.v, (short)0, acc[m], false, false);
    }

    const int po = (y << 7) + x;
#pragma unroll
    for (int m = 0; m < 4; ++m)
#pragma unroll
        for (int r2 = 0; r2 < 8; ++r2) {
            int ch = m * 16 + r2 + 8 * hl;
            xout[((b * 67 + ch) << 14) + po] = acc[m][r2] + bias[ch];
        }
}

// Global average pool of xfin [4,3,H,W] -> pool[12].
__global__ void pool_kernel(const float* __restrict__ xfin, float* __restrict__ pool) {
    __shared__ float red[256];
    int bc = blockIdx.x;
    float s = 0.f;
    for (int i = threadIdx.x; i < HWc; i += 256) s += xfin[bc * HWc + i];
    red[threadIdx.x] = s;
    __syncthreads();
    for (int st = 128; st > 0; st >>= 1) {
        if (threadIdx.x < st) red[threadIdx.x] += red[threadIdx.x + st];
        __syncthreads();
    }
    if (threadIdx.x == 0) pool[bc] = red[0] * (1.f / (float)HWc);
}

// Per-channel dynamic filter branch: chf[b][s][c*9+k].
__global__ void chfilt_kernel(const float* __restrict__ pool,
                              const float* __restrict__ w1, const float* __restrict__ b1,
                              const float* __restrict__ w2, const float* __restrict__ b2,
                              float* __restrict__ chf) {
    int t = blockIdx.x * 256 + threadIdx.x;
    if (t >= 4 * 16 * 27) return;
    int b = t / (16 * 27);
    int r = t - b * 16 * 27;
    int s = r / 27;
    int k = r - s * 27;
    float acc = b2[s * 27 + k];
#pragma unroll
    for (int m = 0; m < 4; ++m) {
        float h = b1[s * 4 + m];
#pragma unroll
        for (int c = 0; c < 3; ++c) h += pool[b * 3 + c] * w1[(s * 4 + m) * 3 + c];
        h = fmaxf(h, 0.f);
        acc += h * w2[(s * 27 + k) * 4 + m];
    }
    chf[(b * 16 + s) * 27 + k] = acc;
}

// Fused DDF filtering + pixel_shuffle(4) + clamp to [0,255].
__global__ __launch_bounds__(256)
void ddf_kernel(const float* __restrict__ xfin, const float* __restrict__ sp,
                const float* __restrict__ chf, float* __restrict__ out) {
    __shared__ float schf[432];
    int t = blockIdx.x * 256 + threadIdx.x;           // one low-res pixel
    int b = t >> 14;
    int r = t & 16383;
    int y = r >> 7;
    int x = r & 127;
    for (int i = threadIdx.x; i < 432; i += 256) schf[i] = chf[b * 432 + i];
    __syncthreads();

    float pch[3][9];
#pragma unroll
    for (int c = 0; c < 3; ++c)
#pragma unroll
        for (int k = 0; k < 9; ++k) {
            int yy = y + k / 3 - 1, xx = x + k % 3 - 1;
            pch[c][k] = ((unsigned)yy < (unsigned)Hc && (unsigned)xx < (unsigned)Wc)
                        ? xfin[((b * 3 + c) << 14) + (yy << 7) + xx] : 0.f;
        }
    for (int s = 0; s < 16; ++s) {
        float spv[9];
#pragma unroll
        for (int k = 0; k < 9; ++k)
            spv[k] = sp[((b * 144 + s * 9 + k) << 14) + r];
        int hy = y * 4 + s / 4, hx = x * 4 + s % 4;
#pragma unroll
        for (int c = 0; c < 3; ++c) {
            float a = 0.f;
#pragma unroll
            for (int k = 0; k < 9; ++k)
                a += pch[c][k] * (schf[s * 27 + c * 9 + k] + spv[k]);
            a = fminf(fmaxf(a, 0.f), 255.f);
            out[((size_t)(b * 3 + c) * 512 + hy) * 512 + hx] = a;
        }
    }
}

// ---------------------------------------------------------------------------
extern "C" void kernel_launch(void* const* d_in, const int* in_sizes, int n_in,
                              void* d_out, int out_size, void* d_ws, size_t ws_size,
                              hipStream_t stream) {
    (void)in_sizes; (void)n_in; (void)out_size; (void)ws_size;
    const float* X      = (const float*)d_in[0];
    const float* lstm_w = (const float*)d_in[1];
    const float* lstm_b = (const float*)d_in[2];
    const float* d_ow[3] = {(const float*)d_in[3],  (const float*)d_in[9],  (const float*)d_in[15]};
    const float* d_ob[3] = {(const float*)d_in[4],  (const float*)d_in[10], (const float*)d_in[16]};
    const float* d_mw[3] = {(const float*)d_in[5],  (const float*)d_in[11], (const float*)d_in[17]};
    const float* d_mb[3] = {(const float*)d_in[6],  (const float*)d_in[12], (const float*)d_in[18]};
    const float* d_w[3]  = {(const float*)d_in[7],  (const float*)d_in[13], (const float*)d_in[19]};
    const float* d_b[3]  = {(const float*)d_in[8],  (const float*)d_in[14], (const float*)d_in[20]};
    const float* conv_w = (const float*)d_in[21];
    const float* conv_b = (const float*)d_in[22];
    const float* sp_w   = (const float*)d_in[23];   // [16,9,3,3,3] == [144,3,3,3]
    const float* sp_b   = (const float*)d_in[24];
    const float* ch_w1  = (const float*)d_in[25];
    const float* ch_b1  = (const float*)d_in[26];
    const float* ch_w2  = (const float*)d_in[27];
    const float* ch_b2  = (const float*)d_in[28];

    float* out  = (float*)d_out;                    // [4,3,512,512]
    float* hid  = out + (size_t)4 * 3 * 512 * 512;  // [4,64,128,128]
    float* cell = hid + (size_t)4 * 64 * HWc;

    // Workspace layout (floats); total ~20.38M floats ~= 81.2 MB.
    float* ws    = (float*)d_ws;
    float* bufA  = ws;                          // [4,67,H,W] = 4390912
    float* bufB  = bufA + 4390912;              // [4,67,H,W]
    float* offb  = bufB + 4390912;              // [4,18,H,W] = 1179648
    float* maskb = offb + 1179648;              // [4, 9,H,W] =  589824
    float* spb   = maskb + 589824;              // [4,144,H,W]= 9437184
    float* xfin  = spb + 9437184;               // [4, 3,H,W] =  196608
    float* pool  = xfin + 196608;               // 16
    float* chf   = pool + 16;                   // 1728
    unsigned short* wpk = (unsigned short*)(chf + 1728);  // 32B-aligned
    unsigned short* w_lstm  = wpk;                          // 16*1*512 =   8192
    unsigned short* w_om[3] = {wpk + 8192,  wpk + 27648,  wpk + 47104};   // 2*19*512 each
    unsigned short* w_dm[3] = {wpk + 66560, wpk + 105472, wpk + 144384};  // 4*19*512 each
    unsigned short* w_fin   = wpk + 183296;                 // 1*19*512
    unsigned short* w_sp    = wpk + 193024;                 // 9*1*512

    dim3 blk(256);

    // Weight fragment packing.
    pack_a<<<32,  blk, 0, stream>>>(lstm_w, 256, nullptr, 0, 67, 3, 16, 1, w_lstm);
    for (int i = 0; i < 3; ++i) {
        pack_a<<<76,  blk, 0, stream>>>(d_ow[i], 18, d_mw[i], 9, 67, 67, 2, 19, w_om[i]);
        pack_a<<<152, blk, 0, stream>>>(d_w[i],  64, nullptr, 0, 67, 67, 4, 19, w_dm[i]);
    }
    pack_a<<<38, blk, 0, stream>>>(conv_w, 3,   nullptr, 0, 67, 67, 1, 19, w_fin);
    pack_a<<<18, blk, 0, stream>>>(sp_w,   144, nullptr, 0, 3,  3,  9, 1,  w_sp);

    set_lr<<<768, blk, 0, stream>>>(X, bufA, bufB);

    // ConvLSTM (h=c=0 -> only X's 3 channels feed the gate conv), fused epilogue.
    conv_wmma<1, 16, 1><<<512, blk, 0, stream>>>(X, 3, w_lstm, lstm_b, nullptr,
                                                 bufA, hid, cell, 0, 0);

    // Three DCNv2 layers, ping-ponging the 67-channel activation buffers.
    float* bin = bufA;
    float* bout = bufB;
    for (int i = 0; i < 3; ++i) {
        conv_wmma<2, 2, 19><<<512, blk, 0, stream>>>(bin, 67, w_om[i],
                                                     d_ob[i], d_mb[i],
                                                     offb, maskb, nullptr, 0, 0);
        deform_wmma<<<512, blk, 0, stream>>>(bin, offb, maskb, w_dm[i], d_b[i], bout);
        float* tmp = bin; bin = bout; bout = tmp;
    }
    // bin now == bufB (layer-3 output + LR channels).

    conv_wmma<0, 1, 19><<<512, blk, 0, stream>>>(bin, 67, w_fin, conv_b, nullptr,
                                                 xfin, nullptr, nullptr, 3, 3);
    conv_wmma<0, 9, 1><<<512, blk, 0, stream>>>(xfin, 3, w_sp, sp_b, nullptr,
                                                spb, nullptr, nullptr, 144, 144);

    pool_kernel<<<12, blk, 0, stream>>>(xfin, pool);
    chfilt_kernel<<<7, blk, 0, stream>>>(pool, ch_w1, ch_b1, ch_w2, ch_b2, chf);
    ddf_kernel<<<256, blk, 0, stream>>>(xfin, spb, chf, out);
}
